// NanoDetPostProcess_12515534700970
// MI455X (gfx1250) — compile-verified, matrix operand order checked by reading, not run on hardware
//
#include <hip/hip_runtime.h>
#include <hip/hip_bf16.h>

// ---------------------------------------------------------------------------
// NanoDet post-process for MI455X (gfx1250, wave32).
//
//   Kernel 1 (decode_boxes<LEVEL>): stream box logits [B,32,H,W] ->
//     async global->LDS stage (ASYNCcnt, global_load_async_to_lds_b32) ->
//     transposed LDS read -> DFL integral via V_WMMA_F32_16X16X32_F16
//     (16 softmax rows x proj vector per WMMA = 4 anchors/wave/step) ->
//     distance2bbox -> boxes[B,8400,4] in ws. Templated on pyramid level so
//     W/HW/stride are compile-time constants (no runtime-magic div/mod).
//   Kernel 2 (nms_kernel): per-image block; LDS histogram (ds_add atomics) of
//     class logits picks an adaptive top-K threshold, gather <=2048 candidates
//     into LDS, then 100 greedy NMS iterations fully in LDS.
//   All normalization divisions use v_rcp_f32 instead of IEEE div sequences.
// ---------------------------------------------------------------------------

typedef __attribute__((ext_vector_type(16))) _Float16 v16h;
typedef __attribute__((ext_vector_type(8)))  float    v8f;

#define IMG_F        640.0f
#define NUM_CLASSES  80
#define BATCH        32
#define A_TOTAL      8400          // 6400 + 1600 + 400
#define TILE         128
#define LOGIT_THR    (-2.9444390f) // log(0.05 / 0.95)  <=> score > 0.05
#define RANGE_HI     5.0f
#define CAND_MAX     2048
#define MAX_NUM      100
#define IOU_THR      0.6f

// ---- async global->LDS helpers (gfx1250) ----------------------------------
#if __has_builtin(__builtin_amdgcn_global_load_async_to_lds_b32)
#define HAVE_ASYNC_LDS 1
typedef __attribute__((address_space(1))) int as1_int;
typedef __attribute__((address_space(3))) int as3_int;
__device__ __forceinline__ void async_g2l_b32(const float* g, float* l) {
    // param0: as1 int* (global src), param1: as3 int* (LDS dst), imm offset, imm cpol
    as1_int* gp = (as1_int*)(uintptr_t)g;
    as3_int* lp = (as3_int*)(unsigned)(uintptr_t)l;  // AS3 ptr is 32-bit offset
    __builtin_amdgcn_global_load_async_to_lds_b32(gp, lp, 0, 0);
}
__device__ __forceinline__ void wait_async_lds() {
#if __has_builtin(__builtin_amdgcn_s_wait_asynccnt)
    __builtin_amdgcn_s_wait_asynccnt(0);
#else
    asm volatile("s_wait_asynccnt 0x0" ::: "memory");
#endif
}
#else
#define HAVE_ASYNC_LDS 0
#endif

// ---------------------------------------------------------------------------
// Kernel 1: DFL integral (WMMA) + bbox decode, templated per pyramid level.
// Block = 128 threads (4 waves); one 128-anchor tile per block.
// ---------------------------------------------------------------------------
template <int LEVEL>
__global__ __launch_bounds__(128) void decode_boxes(
    const float* __restrict__ src, float* __restrict__ boxesOut)
{
    constexpr int   W         = (LEVEL == 0) ? 80 : (LEVEL == 1) ? 40 : 20;
    constexpr int   HW        = W * W;
    constexpr int   levelBase = (LEVEL == 0) ? 0 : (LEVEL == 1) ? 6400 : 8000;
    constexpr float strideF   = (LEVEL == 0) ? 8.f : (LEVEL == 1) ? 16.f : 32.f;
    constexpr int   NT        = (HW + TILE - 1) / TILE;

    __shared__ float sbox[32][TILE + 4]; // +4 pad: break 64-bank conflicts

    const int b    = blockIdx.x / NT;
    const int tile = blockIdx.x % NT;
    const int p0   = tile * TILE;
    const float* base = src + (size_t)b * 32u * (size_t)HW;

    // Stage tile: coalesced global reads (lane = anchor), channel-major [ch][a] LDS write.
    for (int ch = 0; ch < 32; ++ch) {
        for (int a = threadIdx.x; a < TILE; a += blockDim.x) {
            const int p = p0 + a;
#if HAVE_ASYNC_LDS
            if (p < HW) async_g2l_b32(base + (size_t)ch * HW + p, &sbox[ch][a]);
            else        sbox[ch][a] = 0.0f;
#else
            sbox[ch][a] = (p < HW) ? base[(size_t)ch * HW + p] : 0.0f;
#endif
        }
    }
#if HAVE_ASYNC_LDS
    wait_async_lds();
#endif
    __syncthreads();

    const int lane = threadIdx.x & 31;
    const int wave = threadIdx.x >> 5;

    // B matrix: projection vector 0..7 broadcast into every column.
    // 16-bit B 32x16 layout: lanes 0-15 hold K=0..7 in elements 0..7 (N=lane);
    // lanes 16-31 hold K=8..15 (zero: our K is only 8 bins).
    v16h Bv;
#pragma unroll
    for (int i = 0; i < 16; ++i) Bv[i] = (_Float16)0.0f;
    if (lane < 16) {
#pragma unroll
        for (int k = 0; k < 8; ++k) Bv[k] = (_Float16)(float)k;
    }

    for (int it = 0; it < 8; ++it) {
        const int aLoc = wave * 32 + it * 4;   // 4 anchors = 16 dist rows

        // A matrix row r (= lane for lanes 0-15): exp(x - max) over 8 bins.
        const int r    = lane & 15;
        const int aL   = aLoc + (r >> 2);
        const int side = r & 3;

        float v[8], m = -1e30f;
#pragma unroll
        for (int k = 0; k < 8; ++k) {
            const float x = sbox[side * 8 + k][aL];
            v[k] = x;
            m = fmaxf(m, x);
        }
        v16h Av;
#pragma unroll
        for (int i = 0; i < 16; ++i) Av[i] = (_Float16)0.0f;
        float den = 0.0f;
#pragma unroll
        for (int k = 0; k < 8; ++k) {
            const float e = __expf(v[k] - m);
            den += e;
            if (lane < 16) Av[k] = (_Float16)e;   // lanes>=16: A stays zero (K pad)
        }

        v8f C;
#pragma unroll
        for (int i = 0; i < 8; ++i) C[i] = 0.0f;

        // D[m,n] = numerator(row m) for every column n.
        v8f D = __builtin_amdgcn_wmma_f32_16x16x32_f16(
            /*neg_a=*/false, Av, /*neg_b=*/false, Bv,
            /*c_mod=*/(short)0, C, /*reuse_a=*/false, /*reuse_b=*/false);

        // Result lanes: 0..7 -> rows 0..7 (element = lane);
        //               16..23 -> rows 8..15 (element = lane-16).
        const int e = lane & 15;
        float num = D[0];
#pragma unroll
        for (int i = 1; i < 8; ++i) num = (e == i) ? D[i] : num;

        // Denominator lives in the A-row owner lane (row index 0..15).
        const float denX = __shfl(den, (lane < 16) ? lane : (lane - 8), 32);

        const bool resLane = (lane < 8) || (lane >= 16 && lane < 24);
        const int  rr      = (lane < 8) ? lane : (lane - 8);  // row 0..15
        const int  aL2     = aLoc + (rr >> 2);
        const int  side2   = rr & 3;
        const int  P       = p0 + aL2;

        // softmax expectation: one v_rcp_f32 instead of IEEE divide sequence
        const float dist  = num * __builtin_amdgcn_rcpf(denX) * strideF;
        const float c0    = 0.5f * (strideF - 1.0f);
        const float cx    = (float)(P % W) * strideF + c0;   // W is constexpr
        const float cy    = (float)(P / W) * strideF + c0;
        const float cbase = (side2 & 1) ? cy : cx;
        const float sgn   = (side2 < 2) ? -1.0f : 1.0f;
        const float coord = fminf(fmaxf(cbase + sgn * dist, 0.0f), IMG_F);

        if (resLane && P < HW)
            boxesOut[((size_t)b * A_TOTAL + levelBase + P) * 4 + side2] = coord;
    }
}

// ---------------------------------------------------------------------------
// Kernel 2: adaptive top-K candidate selection + greedy NMS, one block/image.
// ---------------------------------------------------------------------------
__global__ __launch_bounds__(256) void nms_kernel(
    const float* __restrict__ cls0, const float* __restrict__ cls1,
    const float* __restrict__ cls2, const float* __restrict__ boxes,
    float* __restrict__ out)
{
    __shared__ int   hist[256];
    __shared__ float cScore[CAND_MAX];
    __shared__ float cBox[CAND_MAX][4];
    __shared__ int   cMeta[CAND_MAX];
    __shared__ int   nCand;
    __shared__ float thrLogit;
    __shared__ float redV[256];
    __shared__ int   redI[256];
    __shared__ float sPick[5];   // [0]=valid flag, [1..4]=picked box
    __shared__ int   sPickCls;

    const int b   = blockIdx.x;
    const int tid = threadIdx.x;

    hist[tid] = 0;
    if (tid == 0) nCand = 0;
    __syncthreads();

    const float* clsArr[3] = {cls0, cls1, cls2};
    const int    HWs[3]    = {6400, 1600, 400};
    const int    lb[3]     = {0, 6400, 8000};
    const float  SCALE     = 256.0f / (RANGE_HI - LOGIT_THR);

    // Pass 1: histogram of logits above the score threshold (LDS ds_add).
    for (int lv = 0; lv < 3; ++lv) {
        const float* c = clsArr[lv] + (size_t)b * NUM_CLASSES * (size_t)HWs[lv];
        const int N = NUM_CLASSES * HWs[lv];
        for (int i = tid; i < N; i += 256) {
            __builtin_prefetch(c + i + 256 * 8, 0, 0);   // global_prefetch_b8
            const float lg = c[i];
            if (lg > LOGIT_THR) {
                int bin = (int)((lg - LOGIT_THR) * SCALE);
                bin = min(bin, 255);
                atomicAdd(&hist[bin], 1);
            }
        }
    }
    __syncthreads();

    // Suffix-scan from the top bin: largest suffix with count <= CAND_MAX.
    if (tid == 0) {
        int acc = 0, bsel = 255;
        for (int i = 255; i >= 0; --i) {
            if (acc + hist[i] > CAND_MAX) break;
            acc += hist[i];
            bsel = i;
        }
        thrLogit = LOGIT_THR + (float)bsel * (RANGE_HI - LOGIT_THR) / 256.0f;
    }
    __syncthreads();
    const float tL = thrLogit;

    // Pass 2: gather candidates above the adaptive threshold into LDS.
    for (int lv = 0; lv < 3; ++lv) {
        const float* c = clsArr[lv] + (size_t)b * NUM_CLASSES * (size_t)HWs[lv];
        const int HW = HWs[lv];
        const int N  = NUM_CLASSES * HW;
        for (int i = tid; i < N; i += 256) {
            const float lg = c[i];
            if (lg >= tL && lg > LOGIT_THR) {
                const int idx = atomicAdd(&nCand, 1);
                if (idx < CAND_MAX) {
                    const int cls = i / HW;
                    const int p   = i - cls * HW;
                    const int aG  = lb[lv] + p;
                    cScore[idx] = __builtin_amdgcn_rcpf(1.0f + __expf(-lg));
                    cMeta[idx]  = (cls << 16) | aG;
                    const float* bx = boxes + ((size_t)b * A_TOTAL + aG) * 4;
                    cBox[idx][0] = bx[0]; cBox[idx][1] = bx[1];
                    cBox[idx][2] = bx[2]; cBox[idx][3] = bx[3];
                }
            }
        }
    }
    __syncthreads();
    const int M = min(nCand, CAND_MAX);

    float* outBoxes  = out;                                   // [B,100,4] f32
    int*   outLabels = (int*)(out + BATCH * MAX_NUM * 4);     // [B,100]   i32
    float* outScores = out + BATCH * MAX_NUM * 4 + BATCH * MAX_NUM; // [B,100]

    for (int it = 0; it < MAX_NUM; ++it) {
        // Block argmax over candidate scores.
        float bv = 0.0f; int bi = -1;
        for (int j = tid; j < M; j += 256) {
            const float s = cScore[j];
            if (s > bv) { bv = s; bi = j; }
        }
        redV[tid] = bv; redI[tid] = bi;
        __syncthreads();
        for (int off = 128; off > 0; off >>= 1) {
            if (tid < off && redV[tid + off] > redV[tid]) {
                redV[tid] = redV[tid + off];
                redI[tid] = redI[tid + off];
            }
            __syncthreads();
        }

        if (tid == 0) {
            const int   pi    = redI[0];
            const float ps    = redV[0];
            const bool  valid = (pi >= 0) && (ps > 0.0f);
            float b0 = 0.f, b1 = 0.f, b2 = 0.f, b3 = 0.f; int cls = -1;
            if (valid) {
                b0 = cBox[pi][0]; b1 = cBox[pi][1];
                b2 = cBox[pi][2]; b3 = cBox[pi][3];
                cls = cMeta[pi] >> 16;
                cScore[pi] = 0.0f;
            }
            const size_t o = (size_t)b * MAX_NUM + it;
            outBoxes[o * 4 + 0] = b0; outBoxes[o * 4 + 1] = b1;
            outBoxes[o * 4 + 2] = b2; outBoxes[o * 4 + 3] = b3;
            outLabels[o] = cls;
            outScores[o] = valid ? ps : 0.0f;
            sPick[0] = valid ? 1.0f : 0.0f;
            sPick[1] = b0; sPick[2] = b1; sPick[3] = b2; sPick[4] = b3;
            sPickCls = cls;
        }
        __syncthreads();

        // Same-class IoU suppression (class-offset trick equivalent).
        if (sPick[0] > 0.0f) {
            const float px1 = sPick[1], py1 = sPick[2];
            const float px2 = sPick[3], py2 = sPick[4];
            const float pa  = (px2 - px1) * (py2 - py1);
            const int   pc  = sPickCls;
            for (int j = tid; j < M; j += 256) {
                if (cScore[j] > 0.0f && (cMeta[j] >> 16) == pc) {
                    const float x1 = fmaxf(px1, cBox[j][0]);
                    const float y1 = fmaxf(py1, cBox[j][1]);
                    const float x2 = fminf(px2, cBox[j][2]);
                    const float y2 = fminf(py2, cBox[j][3]);
                    const float w  = fmaxf(x2 - x1, 0.0f);
                    const float h  = fmaxf(y2 - y1, 0.0f);
                    const float inter = w * h;
                    const float a2 = (cBox[j][2] - cBox[j][0]) * (cBox[j][3] - cBox[j][1]);
                    const float iou = inter * __builtin_amdgcn_rcpf(fmaxf(pa + a2 - inter, 1e-6f));
                    if (iou > IOU_THR) cScore[j] = 0.0f;
                }
            }
        }
        __syncthreads();
    }
}

// ---------------------------------------------------------------------------
// Launch. setup_inputs() dict order: cls0, box0, cls1, box1, cls2, box2.
// Workspace: boxes[B,8400,4] f32 = 4.3 MB.
// Output: boxes[32,100,4] f32 | labels[32,100] i32 | scores[32,100] f32.
// ---------------------------------------------------------------------------
extern "C" void kernel_launch(void* const* d_in, const int* in_sizes, int n_in,
                              void* d_out, int out_size, void* d_ws, size_t ws_size,
                              hipStream_t stream) {
    (void)in_sizes; (void)n_in; (void)out_size; (void)ws_size;
    const float* cls0 = (const float*)d_in[0];
    const float* box0 = (const float*)d_in[1];
    const float* cls1 = (const float*)d_in[2];
    const float* box1 = (const float*)d_in[3];
    const float* cls2 = (const float*)d_in[4];
    const float* box2 = (const float*)d_in[5];

    float* boxesWS = (float*)d_ws;

    decode_boxes<0><<<BATCH * 50, 128, 0, stream>>>(box0, boxesWS); // 6400/128
    decode_boxes<1><<<BATCH * 13, 128, 0, stream>>>(box1, boxesWS); // 1600/128
    decode_boxes<2><<<BATCH * 4,  128, 0, stream>>>(box2, boxesWS); //  400/128
    nms_kernel<<<BATCH, 256, 0, stream>>>(cls0, cls1, cls2, boxesWS, (float*)d_out);
}
